// DualPrompt_69458211110971
// MI455X (gfx1250) — matrix operand order repaired and checked by reading, not compiled, not run on hardware
//
#include <hip/hip_runtime.h>

typedef __attribute__((ext_vector_type(16))) _Float16 v16h;
typedef __attribute__((ext_vector_type(8)))  _Float16 v8h;
typedef __attribute__((ext_vector_type(8)))  float    v8f;
typedef __attribute__((ext_vector_type(4)))  float    v4f;

#define KDIM  768
#define QROWS 2048
#define POOL  4096
#define EPLEN 8
#define XBLK_ELEMS (2048L*197L*768L)
#define EK_ELEMS   (2048L*4L*768L)

// ---------------------------------------------------------------------------
// Row-wise L2 normalize (fp32 in) -> f16 out. One wave32 per row.
// ---------------------------------------------------------------------------
__global__ void DualPrompt_nrm_kernel(const float* __restrict__ in,
                                      _Float16* __restrict__ out, int rows) {
    int row  = blockIdx.x * (blockDim.x >> 5) + (threadIdx.x >> 5);
    int lane = threadIdx.x & 31;
    if (row >= rows) return;
    const float* p = in + (size_t)row * KDIM;
    float ss = 0.f;
    #pragma unroll
    for (int j = 0; j < KDIM / 32; ++j) {
        float v = p[lane + 32 * j];
        ss += v * v;
    }
    #pragma unroll
    for (int m = 16; m >= 1; m >>= 1) ss += __shfl_xor(ss, m, 32);
    float s = 1.0f / fmaxf(sqrtf(ss), 1e-12f);
    _Float16* o = out + (size_t)row * KDIM;
    #pragma unroll
    for (int j = 0; j < KDIM / 32; ++j)
        o[lane + 32 * j] = (_Float16)(p[lane + 32 * j] * s);
}

// ---------------------------------------------------------------------------
// Fused cos-sim GEMM (WMMA f32_16x16x32_f16) + row-wise top-1.
// Block = 256 threads (8 waves). Each block owns 16 query rows, sweeps all
// 4096 pool columns. A tile staged in LDS; B read from L2-resident keys.
// Fragment per-lane layouts follow CDNA5 ISA §7.12.2.
// ---------------------------------------------------------------------------
__global__ __launch_bounds__(256) void DualPrompt_simtop_kernel(
        const _Float16* __restrict__ qn, const _Float16* __restrict__ kn,
        int* __restrict__ kidx) {
    __shared__ _Float16 ldsA[16 * KDIM];       // 24 KB
    __shared__ float    rval[8][16];
    __shared__ int      ridx[8][16];

    const int m0  = blockIdx.x * 16;
    const int tid = threadIdx.x;

    // Stage the 16xK A tile into LDS (16*768 halfs = 1536 v8h chunks).
    {
        const v8h* src = (const v8h*)(qn + (size_t)m0 * KDIM);
        v8h* dst = (v8h*)ldsA;
        #pragma unroll
        for (int j = 0; j < 6; ++j) dst[tid + 256 * j] = src[tid + 256 * j];
    }
    __syncthreads();

    const int wave = tid >> 5;
    const int lane = tid & 31;
    const int hf   = lane >> 4;   // which half of the wave
    const int l16  = lane & 15;

    float best[8];
    int   bidx[8];
    #pragma unroll
    for (int r = 0; r < 8; ++r) { best[r] = -2.0f; bidx[r] = 0; }

    // Each wave sweeps a contiguous range of 32 column tiles (512 columns).
    for (int t = 0; t < 32; ++t) {
        const int n0 = (wave * 32 + t) * 16;
        // B fragment: lane holds column n0+l16; K-dim contiguous in memory.
        const _Float16* bbase = kn + (size_t)(n0 + l16) * KDIM + hf * 8;
        v8f c = {};
        #pragma unroll
        for (int k0 = 0; k0 < KDIM; k0 += 32) {
            // A fragment: lanes 0-15 K = {k0+hf*8..+7, k0+16+hf*8..+7}
            v8h alo = *(const v8h*)(ldsA + l16 * KDIM + k0 + hf * 8);
            v8h ahi = *(const v8h*)(ldsA + l16 * KDIM + k0 + 16 + hf * 8);
            v16h a;
            #pragma unroll
            for (int i = 0; i < 8; ++i) { a[i] = alo[i]; a[8 + i] = ahi[i]; }
            v8h blo = *(const v8h*)(bbase + k0);
            v8h bhi = *(const v8h*)(bbase + k0 + 16);
            v16h b;
            #pragma unroll
            for (int i = 0; i < 8; ++i) { b[i] = blo[i]; b[8 + i] = bhi[i]; }
            c = __builtin_amdgcn_wmma_f32_16x16x32_f16(
                    false, a, false, b, (short)0, c, false, false);
        }
        // c[r] = cos_sim[m0 + hf*8 + r][n0 + l16]
        const int col = n0 + l16;
        #pragma unroll
        for (int r = 0; r < 8; ++r) {
            if (c[r] > best[r]) { best[r] = c[r]; bidx[r] = col; }
        }
    }

    // Reduce across the 16 columns held by each half-wave (tie -> lower idx).
    #pragma unroll
    for (int m = 1; m < 16; m <<= 1) {
        #pragma unroll
        for (int r = 0; r < 8; ++r) {
            float ov = __shfl_xor(best[r], m, 32);
            int   oi = __shfl_xor(bidx[r], m, 32);
            if (ov > best[r] || (ov == best[r] && oi < bidx[r])) {
                best[r] = ov; bidx[r] = oi;
            }
        }
    }
    if (l16 == 0) {
        #pragma unroll
        for (int r = 0; r < 8; ++r) {
            rval[wave][hf * 8 + r] = best[r];
            ridx[wave][hf * 8 + r] = bidx[r];
        }
    }
    __syncthreads();

    // Cross-wave reduction: thread per row.
    if (tid < 16) {
        float bv = rval[0][tid];
        int   bi = ridx[0][tid];
        #pragma unroll
        for (int w = 1; w < 8; ++w) {
            float v = rval[w][tid];
            int   i2 = ridx[w][tid];
            if (v > bv || (v == bv && i2 < bi)) { bv = v; bi = i2; }
        }
        kidx[m0 + tid] = bi;
    }
}

// ---------------------------------------------------------------------------
// Gather e_p[k_idx] and split into Ek (rows 0..3) / Ev (rows 4..7).
// ---------------------------------------------------------------------------
__global__ void DualPrompt_gather_kernel(const float* __restrict__ ep,
                                         const int* __restrict__ kidx,
                                         float* __restrict__ outEk,
                                         float* __restrict__ outEv) {
    const int b   = blockIdx.x;
    const int idx = kidx[b];
    const v4f* src = (const v4f*)(ep + (size_t)idx * EPLEN * KDIM);
    v4f* dk = (v4f*)(outEk + (size_t)b * 4 * KDIM);
    v4f* dv = (v4f*)(outEv + (size_t)b * 4 * KDIM);
    // 4*768/4 = 768 float4 per half
    for (int j = threadIdx.x; j < 768; j += blockDim.x) {
        dk[j] = src[j];
        dv[j] = src[768 + j];
    }
}

// ---------------------------------------------------------------------------
// Streaming passthrough of x_block; nontemporal so the 2.5 GB stream does
// not evict the L2-resident GEMM operands.
// ---------------------------------------------------------------------------
__global__ void DualPrompt_copy_kernel(const float* __restrict__ in,
                                       float* __restrict__ out, long n4) {
    long i      = (long)blockIdx.x * blockDim.x + threadIdx.x;
    long stride = (long)gridDim.x * blockDim.x;
    const v4f* s = (const v4f*)in;
    v4f* d = (v4f*)out;
    for (; i < n4; i += stride) {
        v4f v = __builtin_nontemporal_load(&s[i]);
        __builtin_nontemporal_store(v, &d[i]);
    }
}

extern "C" void kernel_launch(void* const* d_in, const int* in_sizes, int n_in,
                              void* d_out, int out_size, void* d_ws, size_t ws_size,
                              hipStream_t stream) {
    const float* x_querry = (const float*)d_in[0];
    const float* x_block  = (const float*)d_in[1];
    const float* e_k      = (const float*)d_in[2];
    const float* e_p      = (const float*)d_in[3];
    float* out = (float*)d_out;

    char* ws = (char*)d_ws;
    _Float16* qn   = (_Float16*)ws;                                   // 3,145,728 B
    _Float16* kn   = (_Float16*)(ws + (size_t)QROWS * KDIM * 2);      // 6,291,456 B
    int*      kidx = (int*)(ws + (size_t)QROWS * KDIM * 2
                               + (size_t)POOL * KDIM * 2);            // 8,192 B

    // 1) Normalize queries and keys -> f16.
    DualPrompt_nrm_kernel<<<QROWS / 8, 256, 0, stream>>>(x_querry, qn, QROWS);
    DualPrompt_nrm_kernel<<<POOL / 8, 256, 0, stream>>>(e_k, kn, POOL);

    // 2) Fused WMMA GEMM + top-1 argmax.
    DualPrompt_simtop_kernel<<<QROWS / 16, 256, 0, stream>>>(qn, kn, kidx);

    // 3) Gather Ek/Ev.
    float* outEk = out;
    float* outEv = out + EK_ELEMS;
    DualPrompt_gather_kernel<<<QROWS, 256, 0, stream>>>(e_p, kidx, outEk, outEv);

    // 4) Stream x_block through.
    float* outXB = out + 2 * EK_ELEMS;
    DualPrompt_copy_kernel<<<4096, 256, 0, stream>>>(x_block, outXB, XBLK_ELEMS / 4);
}
